// HydModelNet_48971217109003
// MI455X (gfx1250) — matrix operand order repaired and checked by reading, not compiled, not run on hardware
//
#include <hip/hip_runtime.h>

#define STEPS 2048
#define BATCH 1024
#define INPUT_DIM 32

typedef __attribute__((ext_vector_type(16))) __bf16 v16bf;
typedef __attribute__((ext_vector_type(8)))  float  v8f;
typedef __attribute__((ext_vector_type(2)))  __bf16 bf2;
typedef __attribute__((ext_vector_type(2)))  float  f2;

union Tile  { v8f   v; float    f[8]; };
union BFrag { v16bf v; unsigned u[8]; };

// pack two floats into one dword of bf16 (RNE) via hardware convert
__device__ __forceinline__ unsigned pkbf(float a, float b) {
    f2 x; x.x = a; x.y = b;
    bf2 r = __builtin_convertvector(x, bf2);
    return __builtin_bit_cast(unsigned, r);
}

// lane i <-> lane i^16 half-wave swap as a single VALU op (v_permlanex16_b32)
__device__ __forceinline__ float xswap16(float x) {
    int u = __builtin_bit_cast(int, x);
    int r = __builtin_amdgcn_permlanex16(u, u, 0x76543210, 0xfedcba98, false, false);
    return __builtin_bit_cast(float, r);
}

// Build a WMMA B-fragment (32x16 bf16, K = one 32-chunk) from two f32 C-tiles
// of the previous transposed GEMM (t0 = features 32j..32j+15, t1 = 32j+16..32j+31).
// C layout: vgpr i, lane l -> feature = 16*tile + i + 8*(l>=16), batch row = l&15.
// B layout: vgpr v, lanes<16 -> K=2v,2v+1 ; lanes>=16 -> K=16+2v,17+2v.
__device__ __forceinline__ BFrag make_bfrag(const Tile& t0, const Tile& t1, bool lolane) {
    BFrag r;
#pragma unroll
    for (int v = 0; v < 4; ++v) {
        float a0 = xswap16(t1.f[2 * v]);
        float a1 = xswap16(t1.f[2 * v + 1]);
        float s0 = lolane ? t0.f[2 * v]     : a0;
        float s1 = lolane ? t0.f[2 * v + 1] : a1;
        r.u[v] = pkbf(s0, s1);
    }
#pragma unroll
    for (int v = 4; v < 8; ++v) {
        int i = 2 * v - 8;
        float a0 = xswap16(t0.f[i]);
        float a1 = xswap16(t0.f[i + 1]);
        float s0 = lolane ? a0 : t1.f[i];
        float s1 = lolane ? a1 : t1.f[i + 1];
        r.u[v] = pkbf(s0, s1);
    }
    return r;
}

#define WMMA_BF16(ACC, A, B) \
    (ACC) = __builtin_amdgcn_wmma_f32_16x16x32_bf16(false, (A), false, (B), (short)0, (ACC), false, false)

__global__ __launch_bounds__(32)
void hyd_scan_kernel(const float* __restrict__ hyd,
                     const float* __restrict__ Wf0, const float* __restrict__ bf0,
                     const float* __restrict__ Wf1, const float* __restrict__ bf1,
                     const float* __restrict__ Win, const float* __restrict__ bin,
                     const float* __restrict__ Wout, const float* __restrict__ bout,
                     float* __restrict__ out)
{
    // Weight A-fragments (A = W^T, bf16, WMMA A layout): 28 frags * 32 lanes * 8 dwords
    __shared__ __align__(32) unsigned ldsW[28][32][8];   // 28 KB
    // Bias accumulator fragments (v8f per lane per tile): 14 tiles
    __shared__ __align__(32) float    ldsB[14][32][8];   // 14 KB

    const int  lane = threadIdx.x & 31;
    const bool lol  = lane < 16;
    const int  brow = lane & 15;

    // ---------------- one-time weight/bias fragment packing ----------------
    {
        const float* Ws[4] = { Wf0, Wf1, Win, Wout };
        const float* Bs[4] = { bf0, bf1, bin, bout };
        const int    Ks[4] = { 40, 64, 64, 64 };   // K = input features of each GEMM
        const int    Ms[4] = { 64, 64,  9, 80 };   // M = output features
        const int    MT[4] = {  4,  4,  1,  5 };   // M tiles of 16
        int f = 0, bt = 0;
        for (int w = 0; w < 4; ++w) {
            const float* W = Ws[w];
            const int K = Ks[w], M = Ms[w];
            for (int mt = 0; mt < MT[w]; ++mt) {
                for (int kf = 0; kf < 2; ++kf) {
                    const int m = 16 * mt + brow;
                    for (int v = 0; v < 8; ++v) {
                        // A-layout K offsets: lanes<16: {0..15}; lanes>=16: {16..31}
                        int koff = lol ? 2 * v
                                       : 2 * v + 16;
                        int k0 = 32 * kf + koff;
                        float a0 = (m < M && k0     < K) ? W[(size_t)k0       * M + m] : 0.f;
                        float a1 = (m < M && k0 + 1 < K) ? W[(size_t)(k0 + 1) * M + m] : 0.f;
                        ldsW[f][lane][v] = pkbf(a0, a1);
                    }
                    ++f;
                }
                for (int i = 0; i < 8; ++i) {
                    int m = 16 * mt + i + (lol ? 0 : 8);
                    ldsB[bt][lane][i] = (m < M) ? Bs[w][m] : 0.f;
                }
                ++bt;
            }
        }
    }
    __syncthreads();
    // frag bases: Wf0: 0..7, Wf1: 8..15, Win: 16..17, Wout: 18..27
    // bias tiles: Wf0: 0..3, Wf1: 4..7, Win: 8, Wout: 9..13

    // ---------------- recurrent state (valid in lanes 0..15) ----------------
    float st[8];
#pragma unroll
    for (int j = 0; j < 8; ++j) st[j] = 0.f;
    st[0] = 1000.f; st[1] = 100.f;

    const size_t rowg = (size_t)blockIdx.x * 16 + brow;

    for (int t = 0; t < STEPS; ++t) {
        // ---- load x_t: lanes<16 cols 0-15, lanes>=16 cols 16-31 of row (lane&15)
        const float* xp = hyd + ((size_t)t * BATCH + rowg) * INPUT_DIM + (lol ? 0 : 16);
        float4 x0 = ((const float4*)xp)[0];
        float4 x1 = ((const float4*)xp)[1];
        float4 x2 = ((const float4*)xp)[2];
        float4 x3 = ((const float4*)xp)[3];
        if (t + 1 < STEPS)
            __builtin_prefetch((const void*)(xp + (size_t)BATCH * INPUT_DIM), 0, 1);
        float xr[16] = { x0.x, x0.y, x0.z, x0.w, x1.x, x1.y, x1.z, x1.w,
                         x2.x, x2.y, x2.z, x2.w, x3.x, x3.y, x3.z, x3.w };
        const float rain = xr[0];  // col 0 (valid in low lanes)

        // ---- B operand for GEMM1: inp^T = [x | 0.01*stores | 0pad], K=64
        BFrag bx0, bx1;
#pragma unroll
        for (int v = 0; v < 8; ++v) bx0.u[v] = pkbf(xr[2 * v], xr[2 * v + 1]);
#pragma unroll
        for (int v = 0; v < 4; ++v)
            bx1.u[v] = lol ? pkbf(0.01f * st[2 * v], 0.01f * st[2 * v + 1]) : 0u;
#pragma unroll
        for (int v = 4; v < 8; ++v) bx1.u[v] = 0u;

        // ---- GEMM1: h1^T = Wf0^T @ inp^T  (+relu)
        Tile h1[4];
#pragma unroll
        for (int mt = 0; mt < 4; ++mt) {
            v8f acc = *reinterpret_cast<const v8f*>(&ldsB[mt][lane][0]);
            v16bf a0 = *reinterpret_cast<const v16bf*>(&ldsW[mt * 2 + 0][lane][0]);
            WMMA_BF16(acc, a0, bx0.v);
            v16bf a1 = *reinterpret_cast<const v16bf*>(&ldsW[mt * 2 + 1][lane][0]);
            WMMA_BF16(acc, a1, bx1.v);
            h1[mt].v = acc;
#pragma unroll
            for (int i = 0; i < 8; ++i) h1[mt].f[i] = (h1[mt].f[i] > 0.f) ? h1[mt].f[i] : 0.f;
        }
        BFrag hb0 = make_bfrag(h1[0], h1[1], lol);
        BFrag hb1 = make_bfrag(h1[2], h1[3], lol);

        // ---- GEMM2: h2^T = Wf1^T @ h1^T  (+relu)
        Tile h2[4];
#pragma unroll
        for (int mt = 0; mt < 4; ++mt) {
            v8f acc = *reinterpret_cast<const v8f*>(&ldsB[4 + mt][lane][0]);
            v16bf a0 = *reinterpret_cast<const v16bf*>(&ldsW[8 + mt * 2 + 0][lane][0]);
            WMMA_BF16(acc, a0, hb0.v);
            v16bf a1 = *reinterpret_cast<const v16bf*>(&ldsW[8 + mt * 2 + 1][lane][0]);
            WMMA_BF16(acc, a1, hb1.v);
            h2[mt].v = acc;
#pragma unroll
            for (int i = 0; i < 8; ++i) h2[mt].f[i] = (h2[mt].f[i] > 0.f) ? h2[mt].f[i] : 0.f;
        }
        BFrag gb0 = make_bfrag(h2[0], h2[1], lol);
        BFrag gb1 = make_bfrag(h2[2], h2[3], lol);

        // ---- GEMM3: softmax logits (9 of 16 rows valid)
        Tile si;
        {
            v8f acc = *reinterpret_cast<const v8f*>(&ldsB[8][lane][0]);
            v16bf a0 = *reinterpret_cast<const v16bf*>(&ldsW[16][lane][0]);
            WMMA_BF16(acc, a0, gb0.v);
            v16bf a1 = *reinterpret_cast<const v16bf*>(&ldsW[17][lane][0]);
            WMMA_BF16(acc, a1, gb1.v);
            si.v = acc;
        }

        // ---- GEMM4: 80 sigmoid logits (5 tiles)
        Tile so[5];
#pragma unroll
        for (int mt = 0; mt < 5; ++mt) {
            v8f acc = *reinterpret_cast<const v8f*>(&ldsB[9 + mt][lane][0]);
            v16bf a0 = *reinterpret_cast<const v16bf*>(&ldsW[18 + mt * 2 + 0][lane][0]);
            WMMA_BF16(acc, a0, gb0.v);
            v16bf a1 = *reinterpret_cast<const v16bf*>(&ldsW[18 + mt * 2 + 1][lane][0]);
            WMMA_BF16(acc, a1, gb1.v);
            so[mt].v = acc;
        }

        // ---- softmax over 9 cols (low lanes hold rows; col 8 fetched from high half)
        float s8 = xswap16(si.f[0]);
        float sc[9];
#pragma unroll
        for (int i = 0; i < 8; ++i) sc[i] = si.f[i];
        sc[8] = s8;
        float mx = sc[0];
#pragma unroll
        for (int i = 1; i < 9; ++i) mx = fmaxf(mx, sc[i]);
        float e[9], sum = 0.f;
#pragma unroll
        for (int i = 0; i < 9; ++i) { e[i] = __expf(sc[i] - mx); sum += e[i]; }
        float rs = rain / sum;
#pragma unroll
        for (int j = 0; j < 8; ++j) st[j] += e[j + 1] * rs;   // stores += a[:,1:]*rain

        // ---- sigmoid (full wave), then gather all 80 cols into low lanes
#pragma unroll
        for (int mt = 0; mt < 5; ++mt)
#pragma unroll
            for (int i = 0; i < 8; ++i)
                so[mt].f[i] = 1.f / (1.f + __expf(-so[mt].f[i]));
        float bb[80];
#pragma unroll
        for (int mt = 0; mt < 5; ++mt)
#pragma unroll
            for (int i = 0; i < 8; ++i) {
                bb[16 * mt + i]     = so[mt].f[i];
                bb[16 * mt + 8 + i] = xswap16(so[mt].f[i]);
            }

        // ---- store routing (low lanes)
#pragma unroll
        for (int d = 0; d < 8; ++d) {
            float fb[8], s = 0.f;
#pragma unroll
            for (int j = 0; j < 8; ++j) fb[j] = bb[8 * d + j] * st[j];
#pragma unroll
            for (int j = 0; j < 8; ++j) s += fb[j];
#pragma unroll
            for (int j = 0; j < 8; ++j) st[j] -= fb[j];
            st[d] += s;
        }
#pragma unroll
        for (int j = 0; j < 8; ++j) st[j] -= bb[64 + j] * st[j];   // escape
        float fsum = 0.f;
#pragma unroll
        for (int j = 0; j < 8; ++j) {
            float fl = bb[72 + j] * st[j];
            st[j] -= fl;
            fsum += fl;
        }

        if (lol) out[(size_t)t * BATCH + rowg] = fsum;
    }
}

extern "C" void kernel_launch(void* const* d_in, const int* in_sizes, int n_in,
                              void* d_out, int out_size, void* d_ws, size_t ws_size,
                              hipStream_t stream) {
    const float* hyd  = (const float*)d_in[0];
    const float* bf0  = (const float*)d_in[2];
    const float* Wf0  = (const float*)d_in[1];
    const float* Wf1  = (const float*)d_in[3];
    const float* bf1  = (const float*)d_in[4];
    const float* Win  = (const float*)d_in[5];
    const float* bin  = (const float*)d_in[6];
    const float* Wout = (const float*)d_in[7];
    const float* bout = (const float*)d_in[8];
    float* out = (float*)d_out;
    hipLaunchKernelGGL(hyd_scan_kernel, dim3(BATCH / 16), dim3(32), 0, stream,
                       hyd, Wf0, bf0, Wf1, bf1, Win, bin, Wout, bout, out);
}
